// GraphTransformerBlock_10436770529968
// MI455X (gfx1250) — matrix operand order repaired and checked by reading, compiled, not run on hardware
//
#include <hip/hip_runtime.h>
#include <hip/hip_bf16.h>

typedef _Float16 half_t;
typedef __attribute__((ext_vector_type(16))) _Float16 v16h;
typedef __attribute__((ext_vector_type(8)))  _Float16 v8h;
typedef __attribute__((ext_vector_type(8)))  float    v8f;

#define NB   2
#define NN   4096
#define DD   128
#define NH   4
#define HDIM 32
#define MM   (NB * NN)          // 8192
#define NWORDS (NN / 32)        // 128 mask words per row
#define QK_SCALE 0.17677669529663687f  // 1/sqrt(32)
#define NEG_BIG (-3.0e38f)

__device__ __forceinline__ v8h ld8(const half_t* p) { return *(const v8h*)p; }

__device__ __forceinline__ v16h cat8(v8h a, v8h b) {
    v16h r;
#pragma unroll
    for (int i = 0; i < 8; ++i) { r[i] = a[i]; r[i + 8] = b[i]; }
    return r;
}

__device__ __forceinline__ v8f wmma16(v16h a, v16h b, v8f c) {
    return __builtin_amdgcn_wmma_f32_16x16x32_f16(false, a, false, b, (short)0, c, false, false);
}

// ---------------------------------------------------------------------------
// 1) Pack (adj + I) > 0 into a bitmask: [B][N][N/32] u32.  One streaming pass.
// ---------------------------------------------------------------------------
__global__ __launch_bounds__(256) void pack_adj(const float* __restrict__ adj,
                                                unsigned* __restrict__ mask) {
    unsigned idx = blockIdx.x * 256u + threadIdx.x;   // < 2*4096*4096
    unsigned k = idx & 4095u;
    unsigned q = (idx >> 12) & 4095u;
    bool keep = (adj[idx] > 0.0f) || (k == q);
    unsigned long long bal = __ballot(keep);          // wave32: low 32 bits
    if ((threadIdx.x & 31u) == 0u) mask[idx >> 5] = (unsigned)bal;
}

// ---------------------------------------------------------------------------
// 2) f32 -> f16 conversion
// ---------------------------------------------------------------------------
__global__ __launch_bounds__(256) void cvt_f16(const float* __restrict__ s,
                                               half_t* __restrict__ d, int n) {
    int i = blockIdx.x * 256 + threadIdx.x;
    if (i < n) d[i] = (half_t)s[i];
}

// ---------------------------------------------------------------------------
// 3) QKV GEMM: [8192,128]x[128,384] (f16 WMMA, f32 acc) with head-scatter
//    epilogue: q->[bh][n][hd] (pre-scaled 1/sqrt(hd)), k->kT[bh][hd][n], v->[bh][n][hd]
// ---------------------------------------------------------------------------
__global__ __launch_bounds__(256) void gemm_qkv(const half_t* __restrict__ xh,
                                                const half_t* __restrict__ wh,
                                                const float* __restrict__ bias,
                                                half_t* __restrict__ qh,
                                                half_t* __restrict__ kTh,
                                                half_t* __restrict__ vh) {
    const int lane = threadIdx.x & 31, wave = threadIdx.x >> 5;
    const int m0 = blockIdx.x * 128 + wave * 16;
    const int n0 = blockIdx.y * 64;
    const int lr = lane & 15, g = lane >> 4;

    v8f acc[4] = {};
#pragma unroll
    for (int kc = 0; kc < 4; ++kc) {
        const half_t* ap = xh + (m0 + lr) * DD + kc * 32 + g * 8;
        v16h a = cat8(ld8(ap), ld8(ap + 16));
#pragma unroll
        for (int nt = 0; nt < 4; ++nt) {
            const half_t* bp = wh + (kc * 32) * 384 + n0 + nt * 16 + lr;
            v16h b;
#pragma unroll
            for (int i = 0; i < 16; ++i) b[i] = bp[((i & 7) + 8 * g + 16 * (i >> 3)) * 384];
            acc[nt] = wmma16(a, b, acc[nt]);
        }
    }
#pragma unroll
    for (int nt = 0; nt < 4; ++nt) {
#pragma unroll
        for (int r = 0; r < 8; ++r) {
            int m = m0 + r + 8 * g;
            int j = n0 + nt * 16 + lr;
            float v = acc[nt][r] + bias[j];
            int bb = m >> 12, nn = m & 4095;
            int part = j >> 7;                 // 0=q 1=k 2=v
            int h = (j >> 5) & 3, hd = j & 31;
            int bh = bb * NH + h;
            if (part == 0)      qh [((bh * NN) + nn) * HDIM + hd] = (half_t)(v * QK_SCALE);
            else if (part == 1) kTh[((bh * HDIM) + hd) * NN + nn] = (half_t)v;
            else                vh [((bh * NN) + nn) * HDIM + hd] = (half_t)v;
        }
    }
}

// ---------------------------------------------------------------------------
// 4) Flash attention: 1 wave = 16 query rows, loop over 128 key-chunks of 32.
//    Score WMMA (q x kT), bitmask -> online softmax, P via LDS -> P x V WMMA.
// ---------------------------------------------------------------------------
__global__ __launch_bounds__(128) void attn_kernel(const half_t* __restrict__ qh,
                                                   const half_t* __restrict__ kTh,
                                                   const half_t* __restrict__ vh,
                                                   const unsigned* __restrict__ mask,
                                                   half_t* __restrict__ attnh) {
    __shared__ half_t lds[4 * 512];      // 1KB per wave: 16x32 f16 P tile
    const int lane = threadIdx.x & 31, wave = threadIdx.x >> 5;
    const int bh = blockIdx.y;           // 0..7
    const int b = bh >> 2, h = bh & 3;
    const int q0 = blockIdx.x * 64 + wave * 16;
    const int lr = lane & 15, g = lane >> 4;

    const half_t* qp = qh + ((size_t)(bh * NN) + q0 + lr) * HDIM + g * 8;
    v16h qa = cat8(ld8(qp), ld8(qp + 16));

    v8f acc0 = {}, acc1 = {};
    float rowmax[8], rowsum[8];
#pragma unroll
    for (int r = 0; r < 8; ++r) { rowmax[r] = NEG_BIG; rowsum[r] = 0.0f; }

    const unsigned* mrow = mask + ((size_t)b * NN + q0) * NWORDS;
    half_t* pb = lds + wave * 512;

    for (int kc = 0; kc < NWORDS; ++kc) {
        const int k0 = kc * 32;
        // kT B-fragments: lane%16 = key column, 16 hd rows each (table order)
        const half_t* kp = kTh + (size_t)(bh * HDIM) * NN + k0 + lr;
        v16h kb0, kb1;
#pragma unroll
        for (int i = 0; i < 16; ++i) {
            size_t o = (size_t)((i & 7) + 8 * g + 16 * (i >> 3)) * NN;
            kb0[i] = kp[o];
            kb1[i] = kp[o + 16];
        }
        v8f s0 = {}, s1 = {};
        s0 = wmma16(qa, kb0, s0);
        s1 = wmma16(qa, kb1, s1);

        unsigned mw[8];
#pragma unroll
        for (int r = 0; r < 8; ++r) mw[r] = mrow[(r + 8 * g) * NWORDS + kc];

        float p0[8], p1[8];
#pragma unroll
        for (int r = 0; r < 8; ++r) {
            bool k0b = (mw[r] >> lr) & 1u;
            bool k1b = (mw[r] >> (16 + lr)) & 1u;
            float a0 = k0b ? s0[r] : NEG_BIG;
            float a1 = k1b ? s1[r] : NEG_BIG;
            float tm = fmaxf(a0, a1);
            tm = fmaxf(tm, __shfl_xor(tm, 1, 16));
            tm = fmaxf(tm, __shfl_xor(tm, 2, 16));
            tm = fmaxf(tm, __shfl_xor(tm, 4, 16));
            tm = fmaxf(tm, __shfl_xor(tm, 8, 16));
            float nm = fmaxf(rowmax[r], tm);
            float fac = __expf(rowmax[r] - nm);
            rowmax[r] = nm;
            p0[r] = k0b ? __expf(s0[r] - nm) : 0.0f;
            p1[r] = k1b ? __expf(s1[r] - nm) : 0.0f;
            float ts = p0[r] + p1[r];
            ts += __shfl_xor(ts, 1, 16);
            ts += __shfl_xor(ts, 2, 16);
            ts += __shfl_xor(ts, 4, 16);
            ts += __shfl_xor(ts, 8, 16);
            rowsum[r] = rowsum[r] * fac + ts;
            acc0[r] *= fac;
            acc1[r] *= fac;
        }

        // C-layout -> LDS -> A-layout transpose of P (same-wave DS is in-order)
#pragma unroll
        for (int r = 0; r < 8; ++r) {
            pb[(r + 8 * g) * 32 + lr]      = (half_t)p0[r];
            pb[(r + 8 * g) * 32 + 16 + lr] = (half_t)p1[r];
        }
        const half_t* pp = pb + lr * 32 + g * 8;
        v16h pa = cat8(ld8(pp), ld8(pp + 16));

        // V B-fragments: lane%16 = hd column, 16 key rows each (table order)
        const half_t* vp = vh + ((size_t)(bh * NN) + k0) * HDIM + lr;
        v16h vb0, vb1;
#pragma unroll
        for (int i = 0; i < 16; ++i) {
            int o = ((i & 7) + 8 * g + 16 * (i >> 3)) * HDIM;
            vb0[i] = vp[o];
            vb1[i] = vp[o + 16];
        }
        acc0 = wmma16(pa, vb0, acc0);
        acc1 = wmma16(pa, vb1, acc1);
    }

#pragma unroll
    for (int r = 0; r < 8; ++r) {
        float inv = 1.0f / rowsum[r];
        size_t row = (size_t)b * NN + q0 + r + 8 * g;
        attnh[row * DD + h * HDIM + lr]      = (half_t)(acc0[r] * inv);
        attnh[row * DD + h * HDIM + 16 + lr] = (half_t)(acc1[r] * inv);
    }
}

// ---------------------------------------------------------------------------
// 5) Generic [8192,128]x[128,128] GEMM + bias (+optional ReLU), f32 or f16 out
// ---------------------------------------------------------------------------
template <bool RELU, bool OUTH>
__global__ __launch_bounds__(256) void gemm128(const half_t* __restrict__ Ah,
                                               const half_t* __restrict__ Wh,
                                               const float* __restrict__ bias,
                                               float* __restrict__ outf,
                                               half_t* __restrict__ outh) {
    const int lane = threadIdx.x & 31, wave = threadIdx.x >> 5;
    const int m0 = blockIdx.x * 128 + wave * 16;
    const int n0 = blockIdx.y * 64;
    const int lr = lane & 15, g = lane >> 4;

    v8f acc[4] = {};
#pragma unroll
    for (int kc = 0; kc < 4; ++kc) {
        const half_t* ap = Ah + (m0 + lr) * DD + kc * 32 + g * 8;
        v16h a = cat8(ld8(ap), ld8(ap + 16));
#pragma unroll
        for (int nt = 0; nt < 4; ++nt) {
            const half_t* bp = Wh + (kc * 32) * DD + n0 + nt * 16 + lr;
            v16h bfrag;
#pragma unroll
            for (int i = 0; i < 16; ++i) bfrag[i] = bp[((i & 7) + 8 * g + 16 * (i >> 3)) * DD];
            acc[nt] = wmma16(a, bfrag, acc[nt]);
        }
    }
#pragma unroll
    for (int nt = 0; nt < 4; ++nt) {
#pragma unroll
        for (int r = 0; r < 8; ++r) {
            int m = m0 + r + 8 * g;
            int j = n0 + nt * 16 + lr;
            float v = acc[nt][r] + bias[j];
            if (RELU) v = fmaxf(v, 0.0f);
            if (OUTH) outh[(size_t)m * DD + j] = (half_t)v;
            else      outf[(size_t)m * DD + j] = v;
        }
    }
}

// ---------------------------------------------------------------------------
// 6) Residual + LayerNorm over D=128: 1 wave per row, 4 elems/lane
// ---------------------------------------------------------------------------
template <bool EMITH>
__global__ __launch_bounds__(256) void lnres(const float* __restrict__ xin,
                                             const float* __restrict__ proj,
                                             const float* __restrict__ gam,
                                             const float* __restrict__ bet,
                                             float* __restrict__ yout,
                                             half_t* __restrict__ yh) {
    const int lane = threadIdx.x & 31, wave = threadIdx.x >> 5;
    const size_t row = (size_t)blockIdx.x * 8 + wave;
    float4 xv = ((const float4*)(xin + row * DD))[lane];
    float4 pv = ((const float4*)(proj + row * DD))[lane];
    float s[4] = { xv.x + pv.x, xv.y + pv.y, xv.z + pv.z, xv.w + pv.w };
    float sum = s[0] + s[1] + s[2] + s[3];
    float sq  = s[0]*s[0] + s[1]*s[1] + s[2]*s[2] + s[3]*s[3];
#pragma unroll
    for (int o = 16; o >= 1; o >>= 1) {
        sum += __shfl_xor(sum, o, 32);
        sq  += __shfl_xor(sq,  o, 32);
    }
    float mu = sum * (1.0f / DD);
    float var = sq * (1.0f / DD) - mu * mu;
    float rs = rsqrtf(var + 1e-5f);
    int c = lane * 4;
#pragma unroll
    for (int i = 0; i < 4; ++i) {
        float yv = (s[i] - mu) * rs * gam[c + i] + bet[c + i];
        yout[row * DD + c + i] = yv;
        if (EMITH) yh[row * DD + c + i] = (half_t)yv;
    }
}

// ---------------------------------------------------------------------------
extern "C" void kernel_launch(void* const* d_in, const int* in_sizes, int n_in,
                              void* d_out, int out_size, void* d_ws, size_t ws_size,
                              hipStream_t stream) {
    const float* adj   = (const float*)d_in[0];
    const float* x     = (const float*)d_in[1];
    const float* w_qkv = (const float*)d_in[2];
    const float* b_qkv = (const float*)d_in[3];
    const float* w_out = (const float*)d_in[4];
    const float* b_out = (const float*)d_in[5];
    const float* g1    = (const float*)d_in[6];
    const float* b1    = (const float*)d_in[7];
    const float* g2    = (const float*)d_in[8];
    const float* b2    = (const float*)d_in[9];
    const float* w_ff1 = (const float*)d_in[10];
    const float* b_ff1 = (const float*)d_in[11];
    const float* w_ff2 = (const float*)d_in[12];
    const float* b_ff2 = (const float*)d_in[13];

    size_t off = 0;
    auto alloc = [&](size_t bytes) -> void* {
        void* p = (char*)d_ws + off;
        off += (bytes + 255) & ~(size_t)255;
        return p;
    };
    unsigned* mask = (unsigned*)alloc((size_t)NB * NN * NWORDS * 4);
    half_t* xh     = (half_t*)alloc((size_t)MM * DD * 2);
    half_t* wqkvh  = (half_t*)alloc((size_t)DD * 384 * 2);
    half_t* wouth  = (half_t*)alloc((size_t)DD * DD * 2);
    half_t* wff1h  = (half_t*)alloc((size_t)DD * DD * 2);
    half_t* wff2h  = (half_t*)alloc((size_t)DD * DD * 2);
    half_t* qh     = (half_t*)alloc((size_t)NB * NH * NN * HDIM * 2);
    half_t* kTh    = (half_t*)alloc((size_t)NB * NH * NN * HDIM * 2);
    half_t* vh     = (half_t*)alloc((size_t)NB * NH * NN * HDIM * 2);
    half_t* attnh  = (half_t*)alloc((size_t)MM * DD * 2);
    float*  proj   = (float*)alloc((size_t)MM * DD * 4);
    float*  yf     = (float*)alloc((size_t)MM * DD * 4);
    half_t* yh     = (half_t*)alloc((size_t)MM * DD * 2);
    half_t* ffh    = (half_t*)alloc((size_t)MM * DD * 2);
    float*  proj2  = (float*)alloc((size_t)MM * DD * 4);

    // 1) mask packing (single streaming pass over 134MB adj)
    pack_adj<<<(NB * NN * NN) / 256, 256, 0, stream>>>(adj, mask);
    // 2) f16 conversions
    cvt_f16<<<(MM * DD + 255) / 256, 256, 0, stream>>>(x, xh, MM * DD);
    cvt_f16<<<(DD * 384 + 255) / 256, 256, 0, stream>>>(w_qkv, wqkvh, DD * 384);
    cvt_f16<<<(DD * DD + 255) / 256, 256, 0, stream>>>(w_out, wouth, DD * DD);
    cvt_f16<<<(DD * DD + 255) / 256, 256, 0, stream>>>(w_ff1, wff1h, DD * DD);
    cvt_f16<<<(DD * DD + 255) / 256, 256, 0, stream>>>(w_ff2, wff2h, DD * DD);
    // 3) QKV projection with head scatter (kT pre-transposed)
    gemm_qkv<<<dim3(MM / 128, 384 / 64), 256, 0, stream>>>(xh, wqkvh, b_qkv, qh, kTh, vh);
    // 4) masked flash attention
    attn_kernel<<<dim3(NN / 64, NB * NH), 128, 0, stream>>>(qh, kTh, vh, mask, attnh);
    // 5) output projection
    gemm128<false, false><<<dim3(MM / 128, DD / 64), 256, 0, stream>>>(attnh, wouth, b_out, proj, nullptr);
    // 6) residual + LN1 (emit f16 copy for FFN)
    lnres<true><<<MM / 8, 256, 0, stream>>>(x, proj, g1, b1, yf, yh);
    // 7) FFN up (ReLU, f16 out)
    gemm128<true, true><<<dim3(MM / 128, DD / 64), 256, 0, stream>>>(yh, wff1h, b_ff1, nullptr, ffh);
    // 8) FFN down
    gemm128<false, false><<<dim3(MM / 128, DD / 64), 256, 0, stream>>>(ffh, wff2h, b_ff2, proj2, nullptr);
    // 9) residual + LN2 -> final f32 output
    lnres<false><<<MM / 8, 256, 0, stream>>>(yf, proj2, g2, b2, (float*)d_out, nullptr);
}
